// MultiHeadAttention_12592844112285
// MI455X (gfx1250) — matrix operand order repaired and checked by reading, not compiled, not run on hardware
//
#include <hip/hip_runtime.h>
#include <hip/hip_bf16.h>
#include <math.h>

// Problem constants
#define BB   2
#define TT   2048
#define DD   1024     // D_IN == D_OUT
#define HH   16
#define HD   64
#define MTOT (BB*TT)  // 4096 rows

typedef __attribute__((ext_vector_type(16))) __bf16 v16bf;
typedef __attribute__((ext_vector_type(8)))  float  v8f;

// ---------------------------------------------------------------------------
// WMMA helpers (CDNA5 gfx1250, wave32). Layouts per ISA 7.12.2:
//  A (16x32 bf16): lane&15 = row M; element j -> k = (j/8)*16 + (lane>>4)*8 + j%8
//  B (32x16 bf16): lane&15 = col N; element j -> k = (j/8)*16 + (lane>>4)*8 + j%8
//  C/D (16x16 f32): element r -> M = r + 8*(lane>>4), N = lane&15
// ---------------------------------------------------------------------------
__device__ __forceinline__ v8f wmma_bf16(v16bf a, v16bf b, v8f c) {
  return __builtin_amdgcn_wmma_f32_16x16x32_bf16(false, a, false, b, (short)0, c,
                                                 false, false);
}

// A-fragment from LDS tile stored row-major [M][k], 16B-aligned rows.
__device__ __forceinline__ v16bf frag_a_ld(const __bf16* base, int stride, int lane) {
  int r = lane & 15, kh = lane >> 4;
  const __bf16* p = base + r * stride + kh * 8;
  union { v16bf v; uint4 q[2]; } u;
  u.q[0] = *reinterpret_cast<const uint4*>(p);        // j=0..7  : k=kh*8+0..7
  u.q[1] = *reinterpret_cast<const uint4*>(p + 16);   // j=8..15 : k=16+kh*8+0..7
  return u.v;
}

// B-fragment from LDS tile stored K-contiguous per output column: layout [N][k].
__device__ __forceinline__ v16bf frag_b_kcontig(const __bf16* base, int stride, int lane) {
  int n = lane & 15, kh = lane >> 4;
  const __bf16* p = base + n * stride + kh * 8;
  union { v16bf v; uint4 q[2]; } u;
  u.q[0] = *reinterpret_cast<const uint4*>(p);        // j=0..7
  u.q[1] = *reinterpret_cast<const uint4*>(p + 16);   // j=8..15
  return u.v;
}

// ---------------------------------------------------------------------------
// CDNA5 async global->LDS copy (ASYNCcnt-tracked; ISA 10.6 / 15.18.3).
// LDS byte address = low 32 bits of the generic pointer (ISA 10.2 aperture).
// GVS form: mem_addr = SADDR(64b sgpr) + VADDR(i32 vgpr) + inst_offset.
// ---------------------------------------------------------------------------
__device__ __forceinline__ void async_cp_b128(unsigned int lds_byte,
                                              const void* sbase,
                                              unsigned int gbyte) {
  asm volatile("global_load_async_to_lds_b128 %0, %1, %2 offset:0"
               :
               : "v"(lds_byte), "v"(gbyte), "s"(sbase)
               : "memory");
}

template <int N>
__device__ __forceinline__ void wait_asynccnt() {
  asm volatile("s_wait_asynccnt %0" ::"n"(N) : "memory");
}

__device__ __forceinline__ unsigned int lds_off(const void* p) {
  return (unsigned int)(unsigned long long)p;  // addr[31:0] == LDS byte offset
}

// ---------------------------------------------------------------------------
// fp32 -> bf16 elementwise convert (for x)
// ---------------------------------------------------------------------------
__global__ void cvt_bf16_kernel(const float* __restrict__ src,
                                __bf16* __restrict__ dst, int n) {
  int i = blockIdx.x * blockDim.x + threadIdx.x;
  if (i < n) dst[i] = (__bf16)src[i];
}

// ---------------------------------------------------------------------------
// fp32 (K x N) -> bf16 transposed (N x K), tiled through LDS (one-time per W)
// ---------------------------------------------------------------------------
__global__ __launch_bounds__(256) void cvt_transpose_kernel(
    const float* __restrict__ W, __bf16* __restrict__ Wt, int K, int N) {
  __shared__ float tile[32][33];
  int kb = blockIdx.y * 32, nb = blockIdx.x * 32;
  int tx = threadIdx.x, ty = threadIdx.y;
#pragma unroll
  for (int i = ty; i < 32; i += 8)
    tile[i][tx] = W[(size_t)(kb + i) * N + nb + tx];
  __syncthreads();
#pragma unroll
  for (int i = ty; i < 32; i += 8)
    Wt[(size_t)(nb + i) * K + kb + tx] = (__bf16)tile[tx][i];
}

// ---------------------------------------------------------------------------
// Tiled WMMA GEMM:  Y(MxN) = A(MxK) @ Wt(NxK)^T
//   MODE 0: Ybf = bf16(acc * scale)
//   MODE 1: Yf  = acc + bias[col]   (fp32 output)
// Block 256 (8 waves), tile 128x256, BK=32, wave tile 64x64 (16 WMMA per 16
// ds_load_b128). Double-buffered LDS filled by async global->LDS copies.
// ---------------------------------------------------------------------------
template <int MODE>
__global__ __launch_bounds__(256) void gemm_bf16_kernel(
    const __bf16* __restrict__ A, const __bf16* __restrict__ Bt,
    __bf16* __restrict__ Ybf, float* __restrict__ Yf,
    const float* __restrict__ bias, int M, int N, int K, float scale) {
  constexpr int BM = 128, BN = 256, BK = 32, LDT = BK + 8;  // LDT=40 (80B rows)
  __shared__ __bf16 As[2][BM][LDT];   // 20.5 KB
  __shared__ __bf16 Bs[2][BN][LDT];   // 41.0 KB

  const int tid = threadIdx.x, lane = tid & 31, w = tid >> 5;
  const int wm = w >> 2, wn = w & 3;              // 2 x 4 wave grid, 64x64 each
  const int m0 = blockIdx.y * BM, n0 = blockIdx.x * BN;

  v8f acc[4][4];
#pragma unroll
  for (int i = 0; i < 4; ++i)
#pragma unroll
    for (int j = 0; j < 4; ++j)
#pragma unroll
      for (int r = 0; r < 8; ++r) acc[i][j][r] = 0.0f;

  // Per tile per thread: A 32B (2 x b128) + B 2 rows x 32B (4 x b128) = 6 ops.
  const int ldrow = tid >> 1, ldkp = (tid & 1) * 16;
  const unsigned int gArow  = (unsigned int)(m0 + ldrow) * (unsigned int)K;
  const unsigned int gBrow0 = (unsigned int)(n0 + ldrow) * (unsigned int)K;
  const unsigned int gBrow1 = (unsigned int)(n0 + ldrow + 128) * (unsigned int)K;

  auto issue_tile = [&](int kt, int buf) {
    const unsigned int ko = (unsigned int)(kt + ldkp);
    unsigned int lA = lds_off(&As[buf][ldrow][ldkp]);
    unsigned int gA = (gArow + ko) * 2u;
    async_cp_b128(lA, A, gA);
    async_cp_b128(lA + 16u, A, gA + 16u);
    unsigned int lB0 = lds_off(&Bs[buf][ldrow][ldkp]);
    unsigned int gB0 = (gBrow0 + ko) * 2u;
    async_cp_b128(lB0, Bt, gB0);
    async_cp_b128(lB0 + 16u, Bt, gB0 + 16u);
    unsigned int lB1 = lds_off(&Bs[buf][ldrow + 128][ldkp]);
    unsigned int gB1 = (gBrow1 + ko) * 2u;
    async_cp_b128(lB1, Bt, gB1);
    async_cp_b128(lB1 + 16u, Bt, gB1 + 16u);
  };

  const int NK = K / BK;
  issue_tile(0, 0);
  for (int it = 0; it < NK; ++it) {
    const int cur = it & 1;
    if (it + 1 < NK) {
      issue_tile((it + 1) * BK, cur ^ 1);
      wait_asynccnt<6>();   // retire current tile's 6 copies; next 6 in flight
    } else {
      wait_asynccnt<0>();
    }
    __syncthreads();

    v16bf af[4];
#pragma unroll
    for (int i = 0; i < 4; ++i)
      af[i] = frag_a_ld(&As[cur][wm * 64 + i * 16][0], LDT, lane);
#pragma unroll
    for (int j = 0; j < 4; ++j) {
      v16bf bfr = frag_b_kcontig(&Bs[cur][wn * 64 + j * 16][0], LDT, lane);
#pragma unroll
      for (int i = 0; i < 4; ++i) acc[i][j] = wmma_bf16(af[i], bfr, acc[i][j]);
    }
    __syncthreads();  // buffer 'cur' free for reuse next-next iteration
  }

  const int lr = lane & 15, hg = lane >> 4;
  const int row_base = m0 + wm * 64, col_base = n0 + wn * 64;
#pragma unroll
  for (int j = 0; j < 4; ++j) {
    const int col = col_base + j * 16 + lr;
    float bv = (MODE == 1) ? bias[col] : 0.0f;
#pragma unroll
    for (int i = 0; i < 4; ++i) {
#pragma unroll
      for (int r = 0; r < 8; ++r) {
        const int row = row_base + i * 16 + hg * 8 + r;
        if (MODE == 0)
          Ybf[(size_t)row * N + col] = (__bf16)(acc[i][j][r] * scale);
        else
          Yf[(size_t)row * N + col] = acc[i][j][r] + bv;
      }
    }
  }
}

// ---------------------------------------------------------------------------
// Causal flash attention. Grid: (T/64, B*H); block 128 (4 waves).
// Q pre-scaled by 1/sqrt(HD). Q/K/V/ctx stored (B*T, 1024), head = col h*64.
// Q A-fragments are hoisted into registers once (loop-invariant); V is staged
// TRANSPOSED in LDS so K and V B-fragments are both 2x ds_load_b128.
// ---------------------------------------------------------------------------
__global__ __launch_bounds__(128) void flash_attn_kernel(
    const __bf16* __restrict__ Q, const __bf16* __restrict__ Km,
    const __bf16* __restrict__ Vm, __bf16* __restrict__ ctx) {
  constexpr int LDT = HD + 8;  // 72
  __shared__ __bf16 Qs[64][LDT];
  __shared__ __bf16 Ks[64][LDT];
  __shared__ __bf16 Vt[64][LDT];  // [d][key]
  __shared__ __bf16 Ps[64][LDT];

  const int tid = threadIdx.x, lane = tid & 31, w = tid >> 5;
  const int lr = lane & 15, hg = lane >> 4;
  const int qb = blockIdx.x;
  const int bh = blockIdx.y, b = bh >> 4, h = bh & 15;
  const size_t rowoff = (size_t)b * TT;
  const int coloff = h * HD;

  // Load Q tile (64 x 64), 64B per thread
  {
    const int row = tid >> 1, dp = (tid & 1) * 32;
    const uint4* gq = reinterpret_cast<const uint4*>(
        Q + (rowoff + qb * 64 + row) * DD + coloff + dp);
    uint4* sq = reinterpret_cast<uint4*>(&Qs[row][dp]);
#pragma unroll
    for (int i = 0; i < 4; ++i) sq[i] = gq[i];
  }
  __syncthreads();

  // Hoist loop-invariant Q A-fragments into registers
  v16bf qf[2];
  qf[0] = frag_a_ld(&Qs[w * 16][0], LDT, lane);
  qf[1] = frag_a_ld(&Qs[w * 16][32], LDT, lane);

  v8f o[4];
  float mrow[8], lrow[8];
#pragma unroll
  for (int nt = 0; nt < 4; ++nt)
#pragma unroll
    for (int r = 0; r < 8; ++r) o[nt][r] = 0.0f;
#pragma unroll
  for (int r = 0; r < 8; ++r) { mrow[r] = -__builtin_inff(); lrow[r] = 0.0f; }

  for (int kb = 0; kb <= qb; ++kb) {
    // Load K block [key][d]; V block transposed to Vt[d][key]
    {
      const int row = tid >> 1, dp = (tid & 1) * 32;
      const size_t g = (rowoff + kb * 64 + row) * (size_t)DD + coloff + dp;
      const uint4* gk = reinterpret_cast<const uint4*>(Km + g);
      uint4* sk = reinterpret_cast<uint4*>(&Ks[row][dp]);
#pragma unroll
      for (int i = 0; i < 4; ++i) sk[i] = gk[i];
      union { uint4 q[4]; __bf16 hlf[32]; } uv;
      const uint4* gv = reinterpret_cast<const uint4*>(Vm + g);
#pragma unroll
      for (int i = 0; i < 4; ++i) uv.q[i] = gv[i];
#pragma unroll
      for (int jj = 0; jj < 32; ++jj) Vt[dp + jj][row] = uv.hlf[jj];
    }
    __syncthreads();

    // S = Q @ K^T  (wave w: 16 queries x 64 keys)
    v8f s[4];
#pragma unroll
    for (int nt = 0; nt < 4; ++nt)
#pragma unroll
      for (int r = 0; r < 8; ++r) s[nt][r] = 0.0f;
#pragma unroll
    for (int ks = 0; ks < 2; ++ks) {
#pragma unroll
      for (int nt = 0; nt < 4; ++nt) {
        v16bf bfr = frag_b_kcontig(&Ks[nt * 16][ks * 32], LDT, lane);
        s[nt] = wmma_bf16(qf[ks], bfr, s[nt]);
      }
    }

    // Causal mask on the diagonal block
    if (kb == qb) {
#pragma unroll
      for (int nt = 0; nt < 4; ++nt) {
        const int key_l = nt * 16 + lr;
#pragma unroll
        for (int r = 0; r < 8; ++r) {
          const int q_l = w * 16 + hg * 8 + r;
          if (key_l > q_l) s[nt][r] = -__builtin_inff();
        }
      }
    }

    // Online softmax: row stats replicated across each 16-lane group
    float rsum[8], scl[8], newm[8];
#pragma unroll
    for (int r = 0; r < 8; ++r) {
      float m = s[0][r];
#pragma unroll
      for (int nt = 1; nt < 4; ++nt) m = fmaxf(m, s[nt][r]);
#pragma unroll
      for (int off = 1; off < 16; off <<= 1) m = fmaxf(m, __shfl_xor(m, off, 32));
      newm[r] = fmaxf(mrow[r], m);
      scl[r] = __expf(mrow[r] - newm[r]);
      rsum[r] = 0.0f;
    }
#pragma unroll
    for (int nt = 0; nt < 4; ++nt)
#pragma unroll
      for (int r = 0; r < 8; ++r) {
        float p = __expf(s[nt][r] - newm[r]);
        s[nt][r] = p;
        rsum[r] += p;
      }
#pragma unroll
    for (int r = 0; r < 8; ++r) {
#pragma unroll
      for (int off = 1; off < 16; off <<= 1) rsum[r] += __shfl_xor(rsum[r], off, 32);
      lrow[r] = lrow[r] * scl[r] + rsum[r];
      mrow[r] = newm[r];
    }
#pragma unroll
    for (int nt = 0; nt < 4; ++nt)
#pragma unroll
      for (int r = 0; r < 8; ++r) o[nt][r] *= scl[r];

    // Re-layout P (C/D layout -> A-fragment layout) through LDS
#pragma unroll
    for (int nt = 0; nt < 4; ++nt)
#pragma unroll
      for (int r = 0; r < 8; ++r)
        Ps[w * 16 + hg * 8 + r][nt * 16 + lr] = (__bf16)s[nt][r];
    __syncthreads();

    // O += P @ V  (V fragment K-contiguous via Vt)
#pragma unroll
    for (int ks = 0; ks < 2; ++ks) {
      v16bf a = frag_a_ld(&Ps[w * 16][ks * 32], LDT, lane);
#pragma unroll
      for (int nt = 0; nt < 4; ++nt) {
        v16bf bfr = frag_b_kcontig(&Vt[nt * 16][ks * 32], LDT, lane);
        o[nt] = wmma_bf16(a, bfr, o[nt]);
      }
    }
    __syncthreads();  // before next key-block overwrites Ks/Vt
  }

  // Normalize and store ctx (bf16, row-major (B*T, 1024))
  float inv[8];
#pragma unroll
  for (int r = 0; r < 8; ++r) inv[r] = 1.0f / lrow[r];
#pragma unroll
  for (int nt = 0; nt < 4; ++nt)
#pragma unroll
    for (int r = 0; r < 8; ++r) {
      const size_t row = rowoff + qb * 64 + w * 16 + hg * 8 + r;
      ctx[row * DD + coloff + nt * 16 + lr] = (__bf16)(o[nt][r] * inv[r]);
    }
}

// ---------------------------------------------------------------------------
// Launch
// ---------------------------------------------------------------------------
extern "C" void kernel_launch(void* const* d_in, const int* in_sizes, int n_in,
                              void* d_out, int out_size, void* d_ws, size_t ws_size,
                              hipStream_t stream) {
  const float* x  = (const float*)d_in[0];
  const float* Wq = (const float*)d_in[1];
  const float* Wk = (const float*)d_in[2];
  const float* Wv = (const float*)d_in[3];
  const float* Wo = (const float*)d_in[4];
  const float* bo = (const float*)d_in[5];
  float* out = (float*)d_out;

  char* ws = (char*)d_ws;
  size_t off = 0;
  __bf16* xb  = (__bf16*)(ws + off); off += (size_t)MTOT * DD * 2;   // 8 MB
  __bf16* wqt = (__bf16*)(ws + off); off += (size_t)DD * DD * 2;
  __bf16* wkt = (__bf16*)(ws + off); off += (size_t)DD * DD * 2;
  __bf16* wvt = (__bf16*)(ws + off); off += (size_t)DD * DD * 2;
  __bf16* wot = (__bf16*)(ws + off); off += (size_t)DD * DD * 2;
  __bf16* Qb  = (__bf16*)(ws + off); off += (size_t)MTOT * DD * 2;
  __bf16* Kb  = (__bf16*)(ws + off); off += (size_t)MTOT * DD * 2;
  __bf16* Vb  = (__bf16*)(ws + off); off += (size_t)MTOT * DD * 2;
  __bf16* Cb  = (__bf16*)(ws + off); off += (size_t)MTOT * DD * 2;   // 48 MB total

  // 1) convert x -> bf16
  const int nx = MTOT * DD;
  cvt_bf16_kernel<<<nx / 256, 256, 0, stream>>>(x, xb, nx);

  // 2) convert + transpose weights -> bf16 [N][K]
  dim3 tg(DD / 32, DD / 32), tb(32, 8);
  cvt_transpose_kernel<<<tg, tb, 0, stream>>>(Wq, wqt, DD, DD);
  cvt_transpose_kernel<<<tg, tb, 0, stream>>>(Wk, wkt, DD, DD);
  cvt_transpose_kernel<<<tg, tb, 0, stream>>>(Wv, wvt, DD, DD);
  cvt_transpose_kernel<<<tg, tb, 0, stream>>>(Wo, wot, DD, DD);

  // 3) QKV projections (WMMA GEMM, async double-buffered); scale folded into Q
  dim3 gg(DD / 256, MTOT / 128);  // (4, 32)
  const float qscale = 0.125f;    // 1/sqrt(64)
  gemm_bf16_kernel<0><<<gg, 256, 0, stream>>>(xb, wqt, Qb, nullptr, nullptr,
                                              MTOT, DD, DD, qscale);
  gemm_bf16_kernel<0><<<gg, 256, 0, stream>>>(xb, wkt, Kb, nullptr, nullptr,
                                              MTOT, DD, DD, 1.0f);
  gemm_bf16_kernel<0><<<gg, 256, 0, stream>>>(xb, wvt, Vb, nullptr, nullptr,
                                              MTOT, DD, DD, 1.0f);

  // 4) causal flash attention
  dim3 fg(TT / 64, BB * HH);  // (32, 32)
  flash_attn_kernel<<<fg, 128, 0, stream>>>(Qb, Kb, Vb, Cb);

  // 5) output projection + bias (fp32 out)
  gemm_bf16_kernel<1><<<gg, 256, 0, stream>>>(Cb, wot, nullptr, out, bo,
                                              MTOT, DD, DD, 1.0f);
}
